// POS_Guided_Softmax_4028679324209
// MI455X (gfx1250) — matrix operand order repaired and checked by reading, not compiled, not run on hardware
//
#include <hip/hip_runtime.h>

#define N_ 2048
#define H_ 1024
#define V_ 50304
#define K_ 16
#define C_ 3144            // V_/K_
#define TC 512             // columns per block tile
#define CT 7               // ceil(C_/TC)
#define MT 32              // samples per tile (2 x 16-row WMMA fragments)
#define ASTRIDE 1032       // 1024 + 8 bf16 pad (row stride = 516 dwords -> 4-bank skew)
#define BTS 40             // B col stride in bf16 (20 dwords -> conflict-free 16-lane reads)
#define NEG_HUGE (-3.0e38f)

typedef __attribute__((ext_vector_type(16))) __bf16 v16bf;
typedef __attribute__((ext_vector_type(4)))  __bf16 v4bf;
typedef __attribute__((ext_vector_type(8)))  float  v8f;
typedef __attribute__((ext_vector_type(4)))  unsigned int v4u;
typedef __attribute__((ext_vector_type(8)))  unsigned int v8u;

__global__ void zero_counts_kernel(int* counts) {
    if (threadIdx.x < K_) counts[threadIdx.x] = 0;
}

__global__ void scatter_kernel(const int* __restrict__ y_pos,
                               int* __restrict__ counts,
                               int* __restrict__ lists) {
    int n = blockIdx.x * blockDim.x + threadIdx.x;
    if (n < N_) {
        int k = y_pos[n];
        int slot = atomicAdd(&counts[k], 1);
        lists[k * N_ + slot] = n;
    }
}

__global__ void cvt_x_kernel(const float* __restrict__ x, __bf16* __restrict__ xb) {
    int i = blockIdx.x * blockDim.x + threadIdx.x;
    if (i < N_ * H_) xb[i] = (__bf16)x[i];
}

// One wave32 per sample: z = x[n] @ cluster_w, log-softmax over K=16.
__global__ void cluster_ll_kernel(const float* __restrict__ x,
                                  const float* __restrict__ cw,
                                  const int* __restrict__ y_pos,
                                  float* __restrict__ cll) {
    int lane = threadIdx.x & 31;
    int n = blockIdx.x * (blockDim.x >> 5) + (threadIdx.x >> 5);
    if (n >= N_) return;
    float acc[K_];
    #pragma unroll
    for (int j = 0; j < K_; ++j) acc[j] = 0.f;
    for (int h = lane; h < H_; h += 32) {
        float xv = x[(size_t)n * H_ + h];
        const float* cwr = cw + (size_t)h * K_;
        #pragma unroll
        for (int j = 0; j < K_; ++j) acc[j] += xv * cwr[j];
    }
    #pragma unroll
    for (int off = 16; off > 0; off >>= 1) {
        #pragma unroll
        for (int j = 0; j < K_; ++j) acc[j] += __shfl_xor(acc[j], off, 32);
    }
    float m = acc[0];
    #pragma unroll
    for (int j = 1; j < K_; ++j) m = fmaxf(m, acc[j]);
    float s = 0.f;
    #pragma unroll
    for (int j = 0; j < K_; ++j) s += __expf(acc[j] - m);
    if (lane == 0) cll[n] = acc[y_pos[n]] - m - logf(s);
}

__device__ __forceinline__ float4 load_row_clamped(const float* p, int colw) {
    int c0 = colw     < C_ ? colw     : C_ - 1;
    int c1 = colw + 1 < C_ ? colw + 1 : C_ - 1;
    int c2 = colw + 2 < C_ ? colw + 2 : C_ - 1;
    int c3 = colw + 3 < C_ ? colw + 3 : C_ - 1;
    return make_float4(p[c0], p[c1], p[c2], p[c3]);
}

__device__ __forceinline__ void store_bcols(__bf16* dst,
                                            float4 r0, float4 r1,
                                            float4 r2, float4 r3) {
    v4bf c0 = {(__bf16)r0.x, (__bf16)r1.x, (__bf16)r2.x, (__bf16)r3.x};
    v4bf c1 = {(__bf16)r0.y, (__bf16)r1.y, (__bf16)r2.y, (__bf16)r3.y};
    v4bf c2 = {(__bf16)r0.z, (__bf16)r1.z, (__bf16)r2.z, (__bf16)r3.z};
    v4bf c3 = {(__bf16)r0.w, (__bf16)r1.w, (__bf16)r2.w, (__bf16)r3.w};
    *(v4bf*)(dst + 0 * BTS) = c0;
    *(v4bf*)(dst + 1 * BTS) = c1;
    *(v4bf*)(dst + 2 * BTS) = c2;
    *(v4bf*)(dst + 3 * BTS) = c3;
}

__device__ __forceinline__ v16bf load_frag32(const char* base, int step) {
    v4u lo = *(const v4u*)base;
    v4u hi = *(const v4u*)(base + step);
    v8u u = {lo[0], lo[1], lo[2], lo[3], hi[0], hi[1], hi[2], hi[3]};
    return __builtin_bit_cast(v16bf, u);
}

// Grid (CT, K): 256 threads = 8 waves; wave w owns col subtiles 4w..4w+3 for
// a [32 samples x 512 cols] logits tile. bf16 WMMA over H=1024, emitting
// per-(col-tile, sample) online-softmax partials + target-logit scatter.
__global__ __launch_bounds__(256)
void tail_gemm_kernel(const __bf16* __restrict__ xb,
                      const float* __restrict__ logits,
                      const int* __restrict__ lists,
                      const int* __restrict__ counts,
                      const int* __restrict__ y,
                      float* __restrict__ pmax,
                      float* __restrict__ psum,
                      float* __restrict__ tgt) {
    __shared__ __align__(16) __bf16 Ash[MT * ASTRIDE];   // row-major sample tile
    __shared__ __align__(16) __bf16 BshT[TC * BTS];      // col-major weight chunk
    __shared__ float wm[8][MT];
    __shared__ float wsum[8][MT];
    __shared__ int n_sh[MT];
    __shared__ int w_sh[MT];

    const int ct   = blockIdx.x;
    const int k    = blockIdx.y;
    const int cnt  = counts[k];
    const int tid  = threadIdx.x;
    const int lane = tid & 31;
    const int wv   = tid >> 5;
    const int hi   = lane >> 4;     // half-wave select
    const int colL = lane & 15;
    const int nst  = (cnt + MT - 1) / MT;
    const bool tail = (ct * TC + TC > C_);   // block-uniform clamp predicate

    const __bf16* arow0 = &Ash[(lane & 15) * ASTRIDE];
    const __bf16* arow1 = &Ash[((lane & 15) + 16) * ASTRIDE];
    const float*  lbase = logits + (size_t)k * C_;

    for (int st = 0; st < nst; ++st) {
        if (tid < MT) {
            int slot = st * MT + tid;
            int n = -1, win = -1;
            if (slot < cnt) { n = lists[k * N_ + slot]; win = y[n] % C_; }
            n_sh[tid] = n; w_sh[tid] = win;
        }
        __syncthreads();
        // Stage A tile: 32 sample rows x 1024 bf16 (zeros for pad rows).
        for (int i = tid; i < MT * 128; i += 256) {
            int row = i >> 7;
            int ch  = (i & 127) << 3;   // bf16 offset, 8 per float4
            float4 v = make_float4(0.f, 0.f, 0.f, 0.f);
            int n = n_sh[row];
            if (n >= 0) v = *(const float4*)(xb + (size_t)n * H_ + ch);
            *(float4*)(&Ash[row * ASTRIDE + ch]) = v;
        }

        v8f accs[2][4];
        {
            v8f z = {0.f, 0.f, 0.f, 0.f, 0.f, 0.f, 0.f, 0.f};
            #pragma unroll
            for (int mt = 0; mt < 2; ++mt)
                #pragma unroll
                for (int t = 0; t < 4; ++t) accs[mt][t] = z;
        }

        for (int h0 = 0; h0 < H_; h0 += 32) {
            __syncthreads();   // A ready (first iter) / previous BshT consumed
            // Stage B chunk transposed: rows h0..h0+31, cols ct*TC..+TC-1.
            // 4x4 register transpose: coalesced float4 row loads -> bf16 ->
            // contiguous 8B column stores. Clamp path only in the tail block.
            if (!tail) {
                #pragma unroll
                for (int it = 0; it < 4; ++it) {
                    int id = tid + it * 256;
                    int tr = id >> 7;           // row tile 0..7
                    int tc = id & 127;          // col tile
                    int colw = ct * TC + tc * 4;
                    const float* p = lbase + (size_t)(h0 + tr * 4) * V_ + colw;
                    float4 r0 = *(const float4*)(p);
                    float4 r1 = *(const float4*)(p + V_);
                    float4 r2 = *(const float4*)(p + 2 * V_);
                    float4 r3 = *(const float4*)(p + 3 * V_);
                    store_bcols(&BshT[(tc * 4) * BTS + tr * 4], r0, r1, r2, r3);
                }
            } else {
                #pragma unroll
                for (int it = 0; it < 4; ++it) {
                    int id = tid + it * 256;
                    int tr = id >> 7;
                    int tc = id & 127;
                    int colw = ct * TC + tc * 4;
                    const float* p = lbase + (size_t)(h0 + tr * 4) * V_;
                    float4 r0 = load_row_clamped(p, colw);
                    float4 r1 = load_row_clamped(p + V_, colw);
                    float4 r2 = load_row_clamped(p + 2 * V_, colw);
                    float4 r3 = load_row_clamped(p + 3 * V_, colw);
                    store_bcols(&BshT[(tc * 4) * BTS + tr * 4], r0, r1, r2, r3);
                }
            }
            __syncthreads();
            // Load all fragments first, then issue 8 back-to-back WMMAs.
            v16bf a0 = load_frag32((const char*)(arow0 + h0 + hi * 8), 32);
            v16bf a1 = load_frag32((const char*)(arow1 + h0 + hi * 8), 32);
            v16bf bfr[4];
            #pragma unroll
            for (int t = 0; t < 4; ++t) {
                int cs = wv * 4 + t;
                bfr[t] = load_frag32(
                    (const char*)(&BshT[(cs * 16 + colL) * BTS + hi * 8]), 32);
            }
            #pragma unroll
            for (int t = 0; t < 4; ++t)
                accs[0][t] = __builtin_amdgcn_wmma_f32_16x16x32_bf16(
                    false, a0, false, bfr[t], (short)0, accs[0][t], false, false);
            #pragma unroll
            for (int t = 0; t < 4; ++t)
                accs[1][t] = __builtin_amdgcn_wmma_f32_16x16x32_bf16(
                    false, a1, false, bfr[t], (short)0, accs[1][t], false, false);
        }

        // Per-row (sample) online softmax stats across this wave's 64 columns.
        float m_[2][8], s_[2][8];
        #pragma unroll
        for (int mt = 0; mt < 2; ++mt)
            #pragma unroll
            for (int r = 0; r < 8; ++r) { m_[mt][r] = NEG_HUGE; s_[mt][r] = 0.f; }

        #pragma unroll
        for (int mt = 0; mt < 2; ++mt) {
            #pragma unroll
            for (int t = 0; t < 4; ++t) {
                int cs = wv * 4 + t;
                int colw = ct * TC + cs * 16 + colL;
                bool valid = colw < C_;
                v8f acc = accs[mt][t];
                // Target-logit scatter (unique (ct, subtile, col) per sample).
                #pragma unroll
                for (int r = 0; r < 8; ++r) {
                    int grow = mt * 16 + hi * 8 + r;
                    if (valid && w_sh[grow] == colw) tgt[n_sh[grow]] = acc[r];
                }
                float rm[8], rs[8];
                #pragma unroll
                for (int r = 0; r < 8; ++r) rm[r] = valid ? (float)acc[r] : NEG_HUGE;
                #pragma unroll
                for (int off = 1; off < 16; off <<= 1) {
                    #pragma unroll
                    for (int r = 0; r < 8; ++r)
                        rm[r] = fmaxf(rm[r], __shfl_xor(rm[r], off, 32));
                }
                #pragma unroll
                for (int r = 0; r < 8; ++r) {
                    float e = __expf((float)acc[r] - rm[r]);
                    rs[r] = valid ? e : 0.f;
                }
                #pragma unroll
                for (int off = 1; off < 16; off <<= 1) {
                    #pragma unroll
                    for (int r = 0; r < 8; ++r)
                        rs[r] += __shfl_xor(rs[r], off, 32);
                }
                #pragma unroll
                for (int r = 0; r < 8; ++r) {
                    float nm = fmaxf(m_[mt][r], rm[r]);
                    s_[mt][r] = s_[mt][r] * __expf(m_[mt][r] - nm)
                              + rs[r] * __expf(rm[r] - nm);
                    m_[mt][r] = nm;
                }
            }
        }
        if (colL == 0) {
            #pragma unroll
            for (int mt = 0; mt < 2; ++mt)
                #pragma unroll
                for (int r = 0; r < 8; ++r) {
                    wm[wv][mt * 16 + hi * 8 + r]   = m_[mt][r];
                    wsum[wv][mt * 16 + hi * 8 + r] = s_[mt][r];
                }
        }
        __syncthreads();
        if (tid < MT) {
            int n = n_sh[tid];
            if (n >= 0) {
                float M = NEG_HUGE, S = 0.f;
                #pragma unroll
                for (int w = 0; w < 8; ++w) {
                    float mm = wm[w][tid], ss = wsum[w][tid];
                    float nm = fmaxf(M, mm);
                    S = S * __expf(M - nm) + ss * __expf(mm - nm);
                    M = nm;
                }
                pmax[ct * N_ + n] = M;
                psum[ct * N_ + n] = S;
            }
        }
        __syncthreads();
    }
}

__global__ void finalize_kernel(const float* __restrict__ cll,
                                const float* __restrict__ tgt,
                                const float* __restrict__ pmax,
                                const float* __restrict__ psum,
                                float* __restrict__ out) {
    int n = blockIdx.x * blockDim.x + threadIdx.x;
    if (n >= N_) return;
    float M = NEG_HUGE, S = 0.f;
    #pragma unroll
    for (int ct = 0; ct < CT; ++ct) {
        float mm = pmax[ct * N_ + n], ss = psum[ct * N_ + n];
        float nm = fmaxf(M, mm);
        S = S * __expf(M - nm) + ss * __expf(mm - nm);
        M = nm;
    }
    float lse = M + logf(S);
    out[n] = -cll[n] - (tgt[n] - lse);
}

extern "C" void kernel_launch(void* const* d_in, const int* in_sizes, int n_in,
                              void* d_out, int out_size, void* d_ws, size_t ws_size,
                              hipStream_t stream) {
    const float* x      = (const float*)d_in[0];
    const int*   y      = (const int*)d_in[1];
    const int*   y_pos  = (const int*)d_in[2];
    // d_in[3] pos2token, d_in[4] token_in_pos_id: contiguous partition, unused.
    const float* cw     = (const float*)d_in[5];
    const float* logits = (const float*)d_in[6];
    float* out = (float*)d_out;

    char* w = (char*)d_ws;
    auto take = [&](size_t bytes) {
        void* p = (void*)w;
        w += (bytes + 255) & ~(size_t)255;
        return p;
    };
    int*    counts = (int*)take(K_ * sizeof(int));
    int*    lists  = (int*)take((size_t)K_ * N_ * sizeof(int));
    float*  cll    = (float*)take(N_ * sizeof(float));
    float*  tgt    = (float*)take(N_ * sizeof(float));
    float*  pmax   = (float*)take((size_t)CT * N_ * sizeof(float));
    float*  psum   = (float*)take((size_t)CT * N_ * sizeof(float));
    __bf16* xb     = (__bf16*)take((size_t)N_ * H_ * sizeof(__bf16));

    zero_counts_kernel<<<dim3(1), dim3(64), 0, stream>>>(counts);
    scatter_kernel<<<dim3((N_ + 255) / 256), dim3(256), 0, stream>>>(y_pos, counts, lists);
    cvt_x_kernel<<<dim3((N_ * H_ + 255) / 256), dim3(256), 0, stream>>>(x, xb);
    cluster_ll_kernel<<<dim3((N_ + 7) / 8), dim3(256), 0, stream>>>(x, cw, y_pos, cll);
    tail_gemm_kernel<<<dim3(CT, K_), dim3(256), 0, stream>>>(
        xb, logits, lists, counts, y, pmax, psum, tgt);
    finalize_kernel<<<dim3((N_ + 255) / 256), dim3(256), 0, stream>>>(
        cll, tgt, pmax, psum, out);
}
